// Attention_79869211836838
// MI455X (gfx1250) — matrix operand order repaired and checked by reading, not compiled, not run on hardware
//
#include <hip/hip_runtime.h>
#include <hip/hip_bf16.h>
#include <stdint.h>

// Multi-head attention forward for MI455X (gfx1250, wave32, WMMA).
// Stages: f32->bf16 convert, QKV GEMM (WMMA bf16), flash attention (WMMA bf16,
// fp32 online softmax), output projection GEMM (WMMA bf16) + bias.
// Round 4: async builtin signature is (v4i* gsrc, __shared__ v4i* ldst, imm, imm).

#define BB 4
#define NN 2048
#define CC 1024
#define HH 16
#define DD 64
#define BNROWS (BB * NN)           // 8192
#define NEGV  (-10000000.0f)

typedef __attribute__((ext_vector_type(8)))  float          v8f;
typedef __attribute__((ext_vector_type(16))) __bf16         v16bf;
typedef __attribute__((ext_vector_type(4)))  unsigned int   v4u;
typedef __attribute__((ext_vector_type(4)))  int            v4i;

union Frag { v4u q[2]; v16bf v; };   // 32 bytes: one 16-bit A/B WMMA operand

#if __has_builtin(__builtin_amdgcn_global_load_async_to_lds_b128) && \
    __has_builtin(__builtin_amdgcn_s_wait_asynccnt)
#define HAS_ASYNC_LDS 1
#else
#define HAS_ASYNC_LDS 0
#endif

#define GPTR(p) ((v4i*)(p))                                      // generic int4*
#define LPTR(p) ((__attribute__((address_space(3))) v4i*)(p))    // LDS int4*

__device__ __forceinline__ unsigned short f32_bf16(float f) {
  unsigned int u = __float_as_uint(f);
  u += 0x7FFFu + ((u >> 16) & 1u);   // round-to-nearest-even
  return (unsigned short)(u >> 16);
}

__device__ __forceinline__ v8f v8f_zero() {
  v8f z;
#pragma unroll
  for (int i = 0; i < 8; ++i) z[i] = 0.0f;
  return z;
}

// ---------------------------------------------------------------- convert ---
__global__ void cvt_bf16_kernel(const float* __restrict__ src,
                                unsigned short* __restrict__ dst, int n) {
  int i = blockIdx.x * blockDim.x + threadIdx.x;
  int stride = gridDim.x * blockDim.x;
  for (; i < n; i += stride) dst[i] = f32_bf16(src[i]);
}

// ------------------------------------------------------- shared GEMM core ---
// 128x128 block tile, 256 threads (8 waves, wave tile 32x64 = 2x4 WMMA tiles),
// K-step 32, double-buffered LDS. A tile is copy-through (async if available);
// B tile is transposed through VGPRs into [col][k] layout so B fragments are
// two 16B reads (k = 8h+s, 16+8h+s) matching the ISA 7.12.2 B-operand lane layout.
__device__ __forceinline__ void gemm128_core(
    const unsigned short* __restrict__ Amat,
    const unsigned short* __restrict__ Bmat,
    int K, int NT, int row0, int col0,
    unsigned short (&lA)[2][128][40],
    unsigned short (&lBt)[2][128][40],
    v8f (&acc)[2][4]) {
  int tid = threadIdx.x, lane = tid & 31, w = tid >> 5;
  int wr = w & 3, wc = w >> 2;
  int lm = lane & 15;
  int akb = (lane & 16) ? 8 : 0;    // A-operand K base (ISA 7.12.2)
  int bkb = (lane & 16) ? 8 : 0;    // B-operand K base

  // per-thread 16B chunk coordinates (2 chunks each for A and B)
  int cA0 = tid * 2,  ar0 = cA0 >> 2, ak0 = cA0 & 3;
  int cA1 = cA0 + 1,  ar1 = cA1 >> 2, ak1 = cA1 & 3;
  int cB0 = tid * 2,  bk0 = cB0 >> 4, bc0 = cB0 & 15;
  int cB1 = cB0 + 1,  bk1 = cB1 >> 4, bc1 = cB1 & 15;

  const int nIter = K >> 5;

  // ---- prologue: stage tile 0 into buffer 0
#if HAS_ASYNC_LDS
  __builtin_amdgcn_global_load_async_to_lds_b128(
      GPTR(&Amat[(size_t)(row0 + ar0) * K + ak0 * 8]),
      LPTR(&lA[0][ar0][ak0 * 8]), 0, 0);
  __builtin_amdgcn_global_load_async_to_lds_b128(
      GPTR(&Amat[(size_t)(row0 + ar1) * K + ak1 * 8]),
      LPTR(&lA[0][ar1][ak1 * 8]), 0, 0);
#else
  {
    v4u a0 = *(const v4u*)&Amat[(size_t)(row0 + ar0) * K + ak0 * 8];
    v4u a1 = *(const v4u*)&Amat[(size_t)(row0 + ar1) * K + ak1 * 8];
    *(v4u*)&lA[0][ar0][ak0 * 8] = a0;
    *(v4u*)&lA[0][ar1][ak1 * 8] = a1;
  }
#endif
  {
    v4u b0 = *(const v4u*)&Bmat[(size_t)bk0 * NT + col0 + bc0 * 8];
    v4u b1 = *(const v4u*)&Bmat[(size_t)bk1 * NT + col0 + bc1 * 8];
    const unsigned short* h0 = (const unsigned short*)&b0;
    const unsigned short* h1 = (const unsigned short*)&b1;
#pragma unroll
    for (int j = 0; j < 8; ++j) {
      lBt[0][bc0 * 8 + j][bk0] = h0[j];
      lBt[0][bc1 * 8 + j][bk1] = h1[j];
    }
  }

  int cur = 0;
  for (int i = 0; i < nIter; ++i) {
#if HAS_ASYNC_LDS
    __builtin_amdgcn_s_wait_asynccnt(0);   // buffer[cur] async copies done
#endif
    __syncthreads();

    int nxt = cur ^ 1;
    int k0n = (i + 1) << 5;
    bool have_next = (i + 1 < nIter);
    v4u b0s, b1s;
#if !HAS_ASYNC_LDS
    v4u a0s, a1s;
#endif
    if (have_next) {
#if HAS_ASYNC_LDS
      __builtin_amdgcn_global_load_async_to_lds_b128(
          GPTR(&Amat[(size_t)(row0 + ar0) * K + k0n + ak0 * 8]),
          LPTR(&lA[nxt][ar0][ak0 * 8]), 0, 0);
      __builtin_amdgcn_global_load_async_to_lds_b128(
          GPTR(&Amat[(size_t)(row0 + ar1) * K + k0n + ak1 * 8]),
          LPTR(&lA[nxt][ar1][ak1 * 8]), 0, 0);
#else
      a0s = *(const v4u*)&Amat[(size_t)(row0 + ar0) * K + k0n + ak0 * 8];
      a1s = *(const v4u*)&Amat[(size_t)(row0 + ar1) * K + k0n + ak1 * 8];
#endif
      b0s = *(const v4u*)&Bmat[(size_t)(k0n + bk0) * NT + col0 + bc0 * 8];
      b1s = *(const v4u*)&Bmat[(size_t)(k0n + bk1) * NT + col0 + bc1 * 8];
      if (i + 2 < nIter) {                 // pull tile i+2 into cache early
        int k0p = (i + 2) << 5;
        __builtin_prefetch(&Amat[(size_t)(row0 + ar0) * K + k0p + ak0 * 8], 0, 1);
        __builtin_prefetch(&Bmat[(size_t)(k0p + bk0) * NT + col0 + bc0 * 8], 0, 1);
      }
    }

    // ---- compute from buffer[cur]
    Frag a[2];
#pragma unroll
    for (int r = 0; r < 2; ++r) {
      int m = wr * 32 + r * 16 + lm;
      a[r].q[0] = *(const v4u*)&lA[cur][m][akb];
      a[r].q[1] = *(const v4u*)&lA[cur][m][16 + akb];
    }
#pragma unroll
    for (int t = 0; t < 4; ++t) {
      Frag bfr;
      int n = wc * 64 + t * 16 + lm;
      bfr.q[0] = *(const v4u*)&lBt[cur][n][bkb];
      bfr.q[1] = *(const v4u*)&lBt[cur][n][bkb + 16];
#pragma unroll
      for (int r = 0; r < 2; ++r)
        acc[r][t] = __builtin_amdgcn_wmma_f32_16x16x32_bf16(
            false, a[r].v, false, bfr.v, (short)0, acc[r][t], false, false);
    }

    // ---- stage tile i+1 into buffer[nxt]
    if (have_next) {
#if !HAS_ASYNC_LDS
      *(v4u*)&lA[nxt][ar0][ak0 * 8] = a0s;
      *(v4u*)&lA[nxt][ar1][ak1 * 8] = a1s;
#endif
      const unsigned short* h0 = (const unsigned short*)&b0s;
      const unsigned short* h1 = (const unsigned short*)&b1s;
#pragma unroll
      for (int j = 0; j < 8; ++j) {
        lBt[nxt][bc0 * 8 + j][bk0] = h0[j];
        lBt[nxt][bc1 * 8 + j][bk1] = h1[j];
      }
    }
    cur = nxt;
  }
}

// ------------------------------------------------------------- QKV GEMM -----
__global__ __launch_bounds__(256)
void qkv_gemm_kernel(const unsigned short* __restrict__ Xh,
                     const unsigned short* __restrict__ Wh,
                     unsigned short* __restrict__ Qh,
                     unsigned short* __restrict__ Kh,
                     unsigned short* __restrict__ Vh) {
  __shared__ alignas(16) unsigned short lA[2][128][40];
  __shared__ alignas(16) unsigned short lBt[2][128][40];
  int lane = threadIdx.x & 31, w = threadIdx.x >> 5;
  int wr = w & 3, wc = w >> 2;
  int lm = lane & 15, rowoff = (lane & 16) ? 8 : 0;
  int row0 = blockIdx.y * 128, col0 = blockIdx.x * 128;

  v8f acc[2][4];
#pragma unroll
  for (int r = 0; r < 2; ++r)
#pragma unroll
    for (int t = 0; t < 4; ++t) acc[r][t] = v8f_zero();

  gemm128_core(Xh, Wh, CC, 3 * CC, row0, col0, lA, lBt, acc);

  const float scale = 0.125f;   // d^-0.5, folded into Q
#pragma unroll
  for (int r = 0; r < 2; ++r)
#pragma unroll
    for (int t = 0; t < 4; ++t) {
      int colg = col0 + wc * 64 + t * 16 + lm;
      int which = colg >> 10;
      int cw = colg & 1023;
      int h = cw >> 6, dd = cw & 63;
      unsigned short* dst = (which == 0) ? Qh : (which == 1) ? Kh : Vh;
      float s = (which == 0) ? scale : 1.0f;
#pragma unroll
      for (int v = 0; v < 8; ++v) {
        int rowg = row0 + wr * 32 + r * 16 + rowoff + v;
        int b = rowg >> 11, n = rowg & 2047;
        dst[((size_t)(b * HH + h) * NN + n) * DD + dd] =
            f32_bf16(acc[r][t][v] * s);
      }
    }
}

// --------------------------------------------------------- flash attention --
// One block = one (b,h) x 64 q-rows; 4 waves, each owns 16 q-rows.
// K-block = 32 columns per iteration, double-buffered K/V tiles.
__global__ __launch_bounds__(128)
void flash_attn_kernel(const unsigned short* __restrict__ Qh,
                       const unsigned short* __restrict__ Kh,
                       const unsigned short* __restrict__ Vh,
                       const int* __restrict__ attn_mask,
                       const int* __restrict__ padding_mask,
                       unsigned short* __restrict__ Oh) {
  __shared__ alignas(16) unsigned short lK[2][32][72];    // [kcol][d]
  __shared__ alignas(16) unsigned short lVt[2][64][40];   // [d][k] (transposed)
  __shared__ alignas(16) unsigned short lP[4][16][40];    // per-wave P tile

  int tid = threadIdx.x, lane = tid & 31, w = tid >> 5;
  int lm = lane & 15;
  int rowoff = (lane & 16) ? 8 : 0;
  int akb = (lane & 16) ? 8 : 0;
  int bkb = (lane & 16) ? 8 : 0;
  int bh = blockIdx.y, b = bh >> 4, h = bh & 15;
  int q0 = blockIdx.x * 64 + w * 16;

  const unsigned short* Kb = Kh + (size_t)bh * NN * DD;
  const unsigned short* Vb = Vh + (size_t)bh * NN * DD;

  // per-thread 16B chunk coordinates for K/V tiles (32x64 = 256 chunks)
  int c0 = tid * 2, kr0 = c0 >> 3, kc0 = c0 & 7;
  int c1 = c0 + 1,  kr1 = c1 >> 3, kc1 = c1 & 7;

  Frag fq[2];                                   // Q rows resident in registers
  {
    const unsigned short* qr = &Qh[((size_t)bh * NN + q0 + lm) * DD];
#pragma unroll
    for (int f = 0; f < 2; ++f) {
      fq[f].q[0] = *(const v4u*)&qr[f * 32 + akb];
      fq[f].q[1] = *(const v4u*)&qr[f * 32 + 16 + akb];
    }
  }

  float mrun[8], lrun[8];
  v8f accO[4];
#pragma unroll
  for (int v = 0; v < 8; ++v) { mrun[v] = -3.0e38f; lrun[v] = 0.0f; }
#pragma unroll
  for (int t = 0; t < 4; ++t) accO[t] = v8f_zero();

  // ---- prologue: stage tile 0 into buffer 0
#if HAS_ASYNC_LDS
  __builtin_amdgcn_global_load_async_to_lds_b128(
      GPTR(&Kb[(size_t)kr0 * DD + kc0 * 8]), LPTR(&lK[0][kr0][kc0 * 8]), 0, 0);
  __builtin_amdgcn_global_load_async_to_lds_b128(
      GPTR(&Kb[(size_t)kr1 * DD + kc1 * 8]), LPTR(&lK[0][kr1][kc1 * 8]), 0, 0);
#else
  {
    v4u k0v = *(const v4u*)&Kb[(size_t)kr0 * DD + kc0 * 8];
    v4u k1v = *(const v4u*)&Kb[(size_t)kr1 * DD + kc1 * 8];
    *(v4u*)&lK[0][kr0][kc0 * 8] = k0v;
    *(v4u*)&lK[0][kr1][kc1 * 8] = k1v;
  }
#endif
  {
    v4u v0 = *(const v4u*)&Vb[(size_t)kr0 * DD + kc0 * 8];
    v4u v1 = *(const v4u*)&Vb[(size_t)kr1 * DD + kc1 * 8];
    const unsigned short* h0 = (const unsigned short*)&v0;
    const unsigned short* h1 = (const unsigned short*)&v1;
#pragma unroll
    for (int j = 0; j < 8; ++j) {
      lVt[0][kc0 * 8 + j][kr0] = h0[j];
      lVt[0][kc1 * 8 + j][kr1] = h1[j];
    }
  }

  int cur = 0;
  const int nIter = NN >> 5;   // 64
  for (int i = 0; i < nIter; ++i) {
    int kb0 = i << 5;
#if HAS_ASYNC_LDS
    __builtin_amdgcn_s_wait_asynccnt(0);
#endif
    __syncthreads();

    int nxt = cur ^ 1;
    bool have_next = (i + 1 < nIter);
    v4u vs0, vs1;
#if !HAS_ASYNC_LDS
    v4u ks0, ks1;
#endif
    if (have_next) {
      int kbn = kb0 + 32;
#if HAS_ASYNC_LDS
      __builtin_amdgcn_global_load_async_to_lds_b128(
          GPTR(&Kb[(size_t)(kbn + kr0) * DD + kc0 * 8]),
          LPTR(&lK[nxt][kr0][kc0 * 8]), 0, 0);
      __builtin_amdgcn_global_load_async_to_lds_b128(
          GPTR(&Kb[(size_t)(kbn + kr1) * DD + kc1 * 8]),
          LPTR(&lK[nxt][kr1][kc1 * 8]), 0, 0);
#else
      ks0 = *(const v4u*)&Kb[(size_t)(kbn + kr0) * DD + kc0 * 8];
      ks1 = *(const v4u*)&Kb[(size_t)(kbn + kr1) * DD + kc1 * 8];
#endif
      vs0 = *(const v4u*)&Vb[(size_t)(kbn + kr0) * DD + kc0 * 8];
      vs1 = *(const v4u*)&Vb[(size_t)(kbn + kr1) * DD + kc1 * 8];
      if (i + 2 < nIter) {
        int kbp = kb0 + 64;
        __builtin_prefetch(&Kb[(size_t)(kbp + kr0) * DD + kc0 * 8], 0, 1);
        __builtin_prefetch(&Vb[(size_t)(kbp + kr0) * DD + kc0 * 8], 0, 1);
      }
    }

    v8f s[2];                                   // S = Q @ K^T (2 tiles of 16 k)
#pragma unroll
    for (int t = 0; t < 2; ++t) {
      s[t] = v8f_zero();
      Frag bkf0, bkf1;
      int kcol = t * 16 + lm;
      bkf0.q[0] = *(const v4u*)&lK[cur][kcol][bkb];
      bkf0.q[1] = *(const v4u*)&lK[cur][kcol][bkb + 16];
      bkf1.q[0] = *(const v4u*)&lK[cur][kcol][32 + bkb];
      bkf1.q[1] = *(const v4u*)&lK[cur][kcol][32 + bkb + 16];
      s[t] = __builtin_amdgcn_wmma_f32_16x16x32_bf16(
          false, fq[0].v, false, bkf0.v, (short)0, s[t], false, false);
      s[t] = __builtin_amdgcn_wmma_f32_16x16x32_bf16(
          false, fq[1].v, false, bkf1.v, (short)0, s[t], false, false);
    }

#pragma unroll
    for (int t = 0; t < 2; ++t) {               // masks (L2-resident)
      int kg = kb0 + t * 16 + lm;
      int pm = padding_mask[b * NN + kg];
#pragma unroll
      for (int v = 0; v < 8; ++v) {
        int qg = q0 + rowoff + v;
        int am = attn_mask[(size_t)qg * NN + kg];
        float sv = s[t][v];
        sv = (am > 0) ? sv : NEGV;
        sv = (pm > 0) ? NEGV : sv;
        s[t][v] = sv;
      }
    }

    float alpha[8];                             // online softmax (row = VGPR idx)
#pragma unroll
    for (int v = 0; v < 8; ++v) {
      float mv = fmaxf(s[0][v], s[1][v]);
      mv = fmaxf(mv, __shfl_xor(mv, 1, 32));
      mv = fmaxf(mv, __shfl_xor(mv, 2, 32));
      mv = fmaxf(mv, __shfl_xor(mv, 4, 32));
      mv = fmaxf(mv, __shfl_xor(mv, 8, 32));
      float mnew = fmaxf(mrun[v], mv);
      alpha[v] = __expf(mrun[v] - mnew);
      float p0 = __expf(s[0][v] - mnew);
      float p1 = __expf(s[1][v] - mnew);
      s[0][v] = p0; s[1][v] = p1;
      float rs = p0 + p1;
      rs += __shfl_xor(rs, 1, 32);
      rs += __shfl_xor(rs, 2, 32);
      rs += __shfl_xor(rs, 4, 32);
      rs += __shfl_xor(rs, 8, 32);
      lrun[v] = lrun[v] * alpha[v] + rs;
      mrun[v] = mnew;
    }
#pragma unroll
    for (int t = 0; t < 4; ++t)
#pragma unroll
      for (int v = 0; v < 8; ++v) accO[t][v] *= alpha[v];

#pragma unroll
    for (int t = 0; t < 2; ++t)                 // P -> bf16, C-layout -> LDS
#pragma unroll
      for (int v = 0; v < 8; ++v)
        lP[w][rowoff + v][t * 16 + lm] = f32_bf16(s[t][v]);

    __builtin_amdgcn_wave_barrier();            // keep DS write->read order

    Frag pa;                                    // P as A-operand
    pa.q[0] = *(const v4u*)&lP[w][lm][akb];
    pa.q[1] = *(const v4u*)&lP[w][lm][16 + akb];
#pragma unroll
    for (int t = 0; t < 4; ++t) {               // O += P @ V
      Frag bv;
      int dcol = t * 16 + lm;
      bv.q[0] = *(const v4u*)&lVt[cur][dcol][bkb];
      bv.q[1] = *(const v4u*)&lVt[cur][dcol][bkb + 16];
      accO[t] = __builtin_amdgcn_wmma_f32_16x16x32_bf16(
          false, pa.v, false, bv.v, (short)0, accO[t], false, false);
    }

    if (have_next) {                            // stage tile i+1
#if !HAS_ASYNC_LDS
      *(v4u*)&lK[nxt][kr0][kc0 * 8] = ks0;
      *(v4u*)&lK[nxt][kr1][kc1 * 8] = ks1;
#endif
      const unsigned short* h0 = (const unsigned short*)&vs0;
      const unsigned short* h1 = (const unsigned short*)&vs1;
#pragma unroll
      for (int j = 0; j < 8; ++j) {
        lVt[nxt][kc0 * 8 + j][kr0] = h0[j];
        lVt[nxt][kc1 * 8 + j][kr1] = h1[j];
      }
    }
    cur = nxt;
  }

#pragma unroll
  for (int v = 0; v < 8; ++v) lrun[v] = 1.0f / lrun[v];
#pragma unroll
  for (int t = 0; t < 4; ++t) {
    int dcol = t * 16 + lm;
#pragma unroll
    for (int v = 0; v < 8; ++v) {
      int qg = q0 + rowoff + v;
      Oh[(size_t)(b * NN + qg) * CC + h * DD + dcol] =
          f32_bf16(accO[t][v] * lrun[v]);
    }
  }
}

// ------------------------------------------------------ output projection ---
__global__ __launch_bounds__(256)
void proj_gemm_kernel(const unsigned short* __restrict__ Ah,
                      const unsigned short* __restrict__ Wh,
                      const float* __restrict__ bias,
                      float* __restrict__ out) {
  __shared__ alignas(16) unsigned short lA[2][128][40];
  __shared__ alignas(16) unsigned short lBt[2][128][40];
  int lane = threadIdx.x & 31, w = threadIdx.x >> 5;
  int wr = w & 3, wc = w >> 2;
  int lm = lane & 15, rowoff = (lane & 16) ? 8 : 0;
  int row0 = blockIdx.y * 128, col0 = blockIdx.x * 128;

  v8f acc[2][4];
#pragma unroll
  for (int r = 0; r < 2; ++r)
#pragma unroll
    for (int t = 0; t < 4; ++t) acc[r][t] = v8f_zero();

  gemm128_core(Ah, Wh, CC, CC, row0, col0, lA, lBt, acc);

#pragma unroll
  for (int r = 0; r < 2; ++r)
#pragma unroll
    for (int t = 0; t < 4; ++t) {
      int colg = col0 + wc * 64 + t * 16 + lm;
      float bv = bias[colg];
#pragma unroll
      for (int v = 0; v < 8; ++v) {
        int rowg = row0 + wr * 32 + r * 16 + rowoff + v;
        out[(size_t)rowg * CC + colg] = acc[r][t][v] + bv;
      }
    }
}

// ----------------------------------------------------------------- launch ---
extern "C" void kernel_launch(void* const* d_in, const int* in_sizes, int n_in,
                              void* d_out, int out_size, void* d_ws, size_t ws_size,
                              hipStream_t stream) {
  (void)in_sizes; (void)n_in; (void)out_size; (void)ws_size;
  const float* X      = (const float*)d_in[0];
  const int*   attn_m = (const int*)d_in[1];
  const int*   pad_m  = (const int*)d_in[2];
  const float* Wqkv   = (const float*)d_in[3];
  const float* Wproj  = (const float*)d_in[4];
  const float* bproj  = (const float*)d_in[5];
  float* out = (float*)d_out;

  char* ws = (char*)d_ws;                           // 88 MiB total
  unsigned short* Xh  = (unsigned short*)(ws);
  unsigned short* Wqh = (unsigned short*)(ws + (size_t)16 * 1024 * 1024);
  unsigned short* Wph = (unsigned short*)(ws + (size_t)22 * 1024 * 1024);
  unsigned short* Qh  = (unsigned short*)(ws + (size_t)24 * 1024 * 1024);
  unsigned short* Kh  = (unsigned short*)(ws + (size_t)40 * 1024 * 1024);
  unsigned short* Vh  = (unsigned short*)(ws + (size_t)56 * 1024 * 1024);
  unsigned short* Oh  = (unsigned short*)(ws + (size_t)72 * 1024 * 1024);

  cvt_bf16_kernel<<<2048, 256, 0, stream>>>(X, Xh, BNROWS * CC);
  cvt_bf16_kernel<<<1024, 256, 0, stream>>>(Wqkv, Wqh, CC * 3 * CC);
  cvt_bf16_kernel<<<512, 256, 0, stream>>>(Wproj, Wph, CC * CC);

  qkv_gemm_kernel<<<dim3(24, 64), 256, 0, stream>>>(Xh, Wqh, Qh, Kh, Vh);
  flash_attn_kernel<<<dim3(32, 64), 128, 0, stream>>>(Qh, Kh, Vh, attn_m, pad_m, Oh);
  proj_gemm_kernel<<<dim3(8, 64), 256, 0, stream>>>(Oh, Wph, bproj, out);
}
